// GAT_69853348102822
// MI455X (gfx1250) — matrix operand order repaired and checked by reading, not compiled, not run on hardware
//
#include <hip/hip_runtime.h>
#include <hip/hip_bf16.h>

typedef __attribute__((ext_vector_type(16))) _Float16 v16h;
typedef __attribute__((ext_vector_type(8)))  float    v8f;

#define N_NODES  50000
#define N_EDGES  1600000
#define E_TOT    (N_EDGES + N_NODES)   // with self loops
#define IN_DIM   128
#define HID      16
#define HEADS    4
#define C1       (HEADS * HID)         // 64
#define OUT_DIM  32
#define NEG_SLOPE 0.2f

// ---------- helpers ----------

// monotonic float<->uint map so unsigned atomicMax == float max
__device__ __forceinline__ unsigned fmap(float f) {
    unsigned b = __float_as_uint(f);
    return (b & 0x80000000u) ? ~b : (b | 0x80000000u);
}
__device__ __forceinline__ float funmap(unsigned u) {
    unsigned b = (u & 0x80000000u) ? (u ^ 0x80000000u) : ~u;
    return __uint_as_float(b);
}
#define FMAP_NEG_INF 0x007FFFFFu  // fmap(-inf)

__device__ __forceinline__ void edge_sd(const long long* __restrict__ ei, int e, int& s, int& d) {
    if (e < N_EDGES) { s = (int)ei[e]; d = (int)ei[(size_t)N_EDGES + e]; }
    else             { s = d = e - N_EDGES; }
}

__device__ __forceinline__ float lrelu(float x) { return x > 0.f ? x : NEG_SLOPE * x; }

// convert 8 consecutive f32 -> f16 hi + f16 residual lo into fragment slots [base, base+8)
__device__ __forceinline__ void cvtA8(const float4* __restrict__ p, v16h& hi, v16h& lo, int base) {
    float4 f0 = p[0], f1 = p[1];
    float v[8] = {f0.x, f0.y, f0.z, f0.w, f1.x, f1.y, f1.z, f1.w};
#pragma unroll
    for (int i = 0; i < 8; ++i) {
        _Float16 h = (_Float16)v[i];
        hi[base + i] = h;
        lo[base + i] = (_Float16)(v[i] - (float)h);
    }
}
__device__ __forceinline__ void cvtB16(const float4* __restrict__ p, v16h& hi, v16h& lo) {
#pragma unroll
    for (int q = 0; q < 4; ++q) {
        float4 f = p[q];
        float v[4] = {f.x, f.y, f.z, f.w};
#pragma unroll
        for (int j = 0; j < 4; ++j) {
            _Float16 h = (_Float16)v[j];
            hi[q * 4 + j] = h;
            lo[q * 4 + j] = (_Float16)(v[j] - (float)h);
        }
    }
}

// ---------- kernels ----------

__global__ void k_fill_u32(unsigned* __restrict__ p, unsigned val, int n) {
    int i = blockIdx.x * blockDim.x + threadIdx.x;
    if (i < n) p[i] = val;
}

// C[M x Nout] = A[M x K] * B[K x Nout], fp32 in/out, split-precision f16 WMMA.
// One wave per 16x16 tile. M % 16 == 0, K % 32 == 0, Nout % 16 == 0 (all exact here).
__global__ void __launch_bounds__(32) k_gemm_wmma(const float* __restrict__ A,
                                                  const float* __restrict__ B,
                                                  float* __restrict__ C,
                                                  int K, int Nout) {
    const int lane = threadIdx.x;
    const int hlf  = lane >> 4;      // 0 or 1
    const int mr   = lane & 15;
    const int mbase = blockIdx.x * 16;
    const int nbase = blockIdx.y * 16;

    v8f acc = {};
    const float* arow = A + (size_t)(mbase + mr) * K;

    for (int kb = 0; kb < K; kb += 32) {
        v16h ahi, alo, bhi, blo;
        // A 16x32 f16 layout: lanes 0-15 -> K {kb+0..7, kb+16..23}; lanes 16-31 -> K {kb+8..15, kb+24..31}
        cvtA8((const float4*)(arow + kb + 8 * hlf),      ahi, alo, 0);
        cvtA8((const float4*)(arow + kb + 16 + 8 * hlf), ahi, alo, 8);
        // B 32x16 f16 layout: lane%16 = N column, halves follow same K order as A
        const float* bcol = B + (size_t)kb * Nout + nbase + mr;
#pragma unroll
        for (int s = 0; s < 16; ++s) {
            int k = (s < 8) ? (8 * hlf + s) : (16 + 8 * hlf + (s - 8));
            float v = bcol[(size_t)k * Nout];
            _Float16 h = (_Float16)v;
            bhi[s] = h;
            blo[s] = (_Float16)(v - (float)h);
        }

        acc = __builtin_amdgcn_wmma_f32_16x16x32_f16(false, ahi, false, bhi, (short)0, acc, false, false);
        acc = __builtin_amdgcn_wmma_f32_16x16x32_f16(false, alo, false, bhi, (short)0, acc, false, false);
        acc = __builtin_amdgcn_wmma_f32_16x16x32_f16(false, ahi, false, blo, (short)0, acc, false, false);
    }
    // D: VGPR r -> row mbase + r + 8*hlf, col nbase + mr
    float* cr = C + (size_t)(mbase + 8 * hlf) * Nout + nbase + mr;
#pragma unroll
    for (int r = 0; r < 8; ++r) cr[(size_t)r * Nout] = acc[r];
}

// per (node, head): attention logits for layer 1
__global__ void k_attn1(const float* __restrict__ h1, const float* __restrict__ a_src,
                        const float* __restrict__ a_dst,
                        float* __restrict__ als, float* __restrict__ ald) {
    int t = blockIdx.x * blockDim.x + threadIdx.x;
    if (t >= N_NODES * HEADS) return;
    int n = t >> 2, hd = t & 3;
    const float4* hp = (const float4*)(h1 + (size_t)n * C1 + hd * HID);
    const float4* sp = (const float4*)(a_src + hd * HID);
    const float4* dp = (const float4*)(a_dst + hd * HID);
    float s = 0.f, d = 0.f;
#pragma unroll
    for (int q = 0; q < 4; ++q) {
        float4 hv = hp[q], sv = sp[q], dv = dp[q];
        s += hv.x * sv.x + hv.y * sv.y + hv.z * sv.z + hv.w * sv.w;
        d += hv.x * dv.x + hv.y * dv.y + hv.z * dv.z + hv.w * dv.w;
    }
    als[t] = s; ald[t] = d;
}

// per node: attention logits for layer 2 (1 head, 32 ch)
__global__ void k_attn2(const float* __restrict__ h2, const float* __restrict__ a_src,
                        const float* __restrict__ a_dst,
                        float* __restrict__ als, float* __restrict__ ald) {
    int n = blockIdx.x * blockDim.x + threadIdx.x;
    if (n >= N_NODES) return;
    const float4* hp = (const float4*)(h2 + (size_t)n * OUT_DIM);
    const float4* sp = (const float4*)a_src;
    const float4* dp = (const float4*)a_dst;
    float s = 0.f, d = 0.f;
#pragma unroll
    for (int q = 0; q < 8; ++q) {
        float4 hv = hp[q], sv = sp[q], dv = dp[q];
        s += hv.x * sv.x + hv.y * sv.y + hv.z * sv.z + hv.w * sv.w;
        d += hv.x * dv.x + hv.y * dv.y + hv.z * dv.z + hv.w * dv.w;
    }
    als[n] = s; ald[n] = d;
}

// layer1 edge pass A: e = lrelu(als[src]+ald[dst]); store e; segment max into m (mapped uint)
__global__ void k_edge_max1(const long long* __restrict__ ei, const float* __restrict__ als,
                            const float* __restrict__ ald, float* __restrict__ ebuf,
                            unsigned* __restrict__ mu) {
    int e = blockIdx.x * blockDim.x + threadIdx.x;
    if (e >= E_TOT) return;
    int s, d; edge_sd(ei, e, s, d);
    float4 as = *(const float4*)(als + (size_t)s * 4);
    float4 ad = *(const float4*)(ald + (size_t)d * 4);
    float4 ev;
    ev.x = lrelu(as.x + ad.x); ev.y = lrelu(as.y + ad.y);
    ev.z = lrelu(as.z + ad.z); ev.w = lrelu(as.w + ad.w);
    *(float4*)(ebuf + (size_t)e * 4) = ev;
    unsigned* mp = mu + (size_t)d * 4;
    atomicMax(mp + 0, fmap(ev.x)); atomicMax(mp + 1, fmap(ev.y));
    atomicMax(mp + 2, fmap(ev.z)); atomicMax(mp + 3, fmap(ev.w));
}

// layer1 edge pass B: ex = exp(e - m[dst]); store ex; Z[dst] += ex
__global__ void k_edge_exp1(const long long* __restrict__ ei, float* __restrict__ ebuf,
                            const unsigned* __restrict__ mu, float* __restrict__ Z) {
    int e = blockIdx.x * blockDim.x + threadIdx.x;
    if (e >= E_TOT) return;
    int s, d; edge_sd(ei, e, s, d);
    float4 ev = *(const float4*)(ebuf + (size_t)e * 4);
    uint4  mv = *(const uint4*)(mu + (size_t)d * 4);
    float4 ex;
    ex.x = expf(ev.x - funmap(mv.x)); ex.y = expf(ev.y - funmap(mv.y));
    ex.z = expf(ev.z - funmap(mv.z)); ex.w = expf(ev.w - funmap(mv.w));
    *(float4*)(ebuf + (size_t)e * 4) = ex;
    float* zp = Z + (size_t)d * 4;
    atomicAdd(zp + 0, ex.x); atomicAdd(zp + 1, ex.y);
    atomicAdd(zp + 2, ex.z); atomicAdd(zp + 3, ex.w);
}

// layer1 message: per (edge, head) thread, 16 channels
__global__ void k_edge_msg1(const long long* __restrict__ ei, const float* __restrict__ ebuf,
                            const float* __restrict__ Z, const float* __restrict__ h1,
                            float* __restrict__ out1) {
    int t = blockIdx.x * blockDim.x + threadIdx.x;
    if (t >= E_TOT * HEADS) return;
    int e = t >> 2, hd = t & 3;
    int s, d; edge_sd(ei, e, s, d);
    float alpha = ebuf[(size_t)e * 4 + hd] / (Z[(size_t)d * 4 + hd] + 1e-16f);
    const float4* hp = (const float4*)(h1 + (size_t)s * C1 + hd * HID);
    float4 v0 = hp[0], v1 = hp[1], v2 = hp[2], v3 = hp[3];
    float* op = out1 + (size_t)d * C1 + hd * HID;
    atomicAdd(op + 0,  v0.x * alpha); atomicAdd(op + 1,  v0.y * alpha);
    atomicAdd(op + 2,  v0.z * alpha); atomicAdd(op + 3,  v0.w * alpha);
    atomicAdd(op + 4,  v1.x * alpha); atomicAdd(op + 5,  v1.y * alpha);
    atomicAdd(op + 6,  v1.z * alpha); atomicAdd(op + 7,  v1.w * alpha);
    atomicAdd(op + 8,  v2.x * alpha); atomicAdd(op + 9,  v2.y * alpha);
    atomicAdd(op + 10, v2.z * alpha); atomicAdd(op + 11, v2.w * alpha);
    atomicAdd(op + 12, v3.x * alpha); atomicAdd(op + 13, v3.y * alpha);
    atomicAdd(op + 14, v3.z * alpha); atomicAdd(op + 15, v3.w * alpha);
}

// layer1 finalize: out1 = elu(out1 + bias1), in place
__global__ void k_fin1(float* __restrict__ out1, const float* __restrict__ bias1) {
    int t = blockIdx.x * blockDim.x + threadIdx.x;
    if (t >= N_NODES * C1) return;
    float v = out1[t] + bias1[t & (C1 - 1)];
    out1[t] = v > 0.f ? v : expm1f(v);
}

// layer2 edge pass A (1 head)
__global__ void k_edge_max2(const long long* __restrict__ ei, const float* __restrict__ als,
                            const float* __restrict__ ald, float* __restrict__ ebuf,
                            unsigned* __restrict__ mu) {
    int e = blockIdx.x * blockDim.x + threadIdx.x;
    if (e >= E_TOT) return;
    int s, d; edge_sd(ei, e, s, d);
    float ev = lrelu(als[s] + ald[d]);
    ebuf[e] = ev;
    atomicMax(mu + d, fmap(ev));
}

__global__ void k_edge_exp2(const long long* __restrict__ ei, float* __restrict__ ebuf,
                            const unsigned* __restrict__ mu, float* __restrict__ Z) {
    int e = blockIdx.x * blockDim.x + threadIdx.x;
    if (e >= E_TOT) return;
    int s, d; edge_sd(ei, e, s, d);
    float ex = expf(ebuf[e] - funmap(mu[d]));
    ebuf[e] = ex;
    atomicAdd(Z + d, ex);
}

__global__ void k_edge_msg2(const long long* __restrict__ ei, const float* __restrict__ ebuf,
                            const float* __restrict__ Z, const float* __restrict__ h2,
                            float* __restrict__ out) {
    int e = blockIdx.x * blockDim.x + threadIdx.x;
    if (e >= E_TOT) return;
    int s, d; edge_sd(ei, e, s, d);
    float alpha = ebuf[e] / (Z[d] + 1e-16f);
    const float4* hp = (const float4*)(h2 + (size_t)s * OUT_DIM);
    float* op = out + (size_t)d * OUT_DIM;
#pragma unroll
    for (int q = 0; q < 8; ++q) {
        float4 v = hp[q];
        atomicAdd(op + q * 4 + 0, v.x * alpha);
        atomicAdd(op + q * 4 + 1, v.y * alpha);
        atomicAdd(op + q * 4 + 2, v.z * alpha);
        atomicAdd(op + q * 4 + 3, v.w * alpha);
    }
}

__global__ void k_bias2(float* __restrict__ out, const float* __restrict__ bias2) {
    int t = blockIdx.x * blockDim.x + threadIdx.x;
    if (t >= N_NODES * OUT_DIM) return;
    out[t] += bias2[t & (OUT_DIM - 1)];
}

// ---------- launcher ----------

extern "C" void kernel_launch(void* const* d_in, const int* in_sizes, int n_in,
                              void* d_out, int out_size, void* d_ws, size_t ws_size,
                              hipStream_t stream) {
    const float*     x    = (const float*)d_in[0];
    const long long* ei   = (const long long*)d_in[1];
    const float*     W1   = (const float*)d_in[2];
    const float*     as1  = (const float*)d_in[3];
    const float*     ad1  = (const float*)d_in[4];
    const float*     b1   = (const float*)d_in[5];
    const float*     W2   = (const float*)d_in[6];
    const float*     as2  = (const float*)d_in[7];
    const float*     ad2  = (const float*)d_in[8];
    const float*     b2   = (const float*)d_in[9];
    float*           out  = (float*)d_out;

    char* w = (char*)d_ws;
    auto alloc = [&](size_t bytes) { void* p = (void*)w; w += (bytes + 255) & ~(size_t)255; return p; };

    float*    h1   = (float*)alloc((size_t)N_NODES * C1 * 4);
    float*    out1 = (float*)alloc((size_t)N_NODES * C1 * 4);   // later reused as h2 input
    float*    h2   = (float*)alloc((size_t)N_NODES * OUT_DIM * 4);
    float*    al_s1= (float*)alloc((size_t)N_NODES * HEADS * 4);
    float*    al_d1= (float*)alloc((size_t)N_NODES * HEADS * 4);
    unsigned* m1u  = (unsigned*)alloc((size_t)N_NODES * HEADS * 4);
    float*    Z1   = (float*)alloc((size_t)N_NODES * HEADS * 4);
    float*    al_s2= (float*)alloc((size_t)N_NODES * 4);
    float*    al_d2= (float*)alloc((size_t)N_NODES * 4);
    unsigned* m2u  = (unsigned*)alloc((size_t)N_NODES * 4);
    float*    Z2   = (float*)alloc((size_t)N_NODES * 4);
    float*    ebuf = (float*)alloc((size_t)E_TOT * HEADS * 4);  // layer2 reuses first E_TOT floats

    const int B = 256;
    auto g = [](long long n, int b) { return (unsigned)((n + b - 1) / b); };

    // ---- layer 1 ----
    k_gemm_wmma<<<dim3(N_NODES / 16, C1 / 16), 32, 0, stream>>>(x, W1, h1, IN_DIM, C1);
    k_attn1<<<g((long long)N_NODES * HEADS, B), B, 0, stream>>>(h1, as1, ad1, al_s1, al_d1);
    k_fill_u32<<<g((long long)N_NODES * HEADS, B), B, 0, stream>>>(m1u, FMAP_NEG_INF, N_NODES * HEADS);
    k_fill_u32<<<g((long long)N_NODES * HEADS, B), B, 0, stream>>>((unsigned*)Z1, 0u, N_NODES * HEADS);
    k_fill_u32<<<g((long long)N_NODES * C1, B), B, 0, stream>>>((unsigned*)out1, 0u, N_NODES * C1);
    k_edge_max1<<<g(E_TOT, B), B, 0, stream>>>(ei, al_s1, al_d1, ebuf, m1u);
    k_edge_exp1<<<g(E_TOT, B), B, 0, stream>>>(ei, ebuf, m1u, Z1);
    k_edge_msg1<<<g((long long)E_TOT * HEADS, B), B, 0, stream>>>(ei, ebuf, Z1, h1, out1);
    k_fin1<<<g((long long)N_NODES * C1, B), B, 0, stream>>>(out1, b1);

    // ---- layer 2 ----
    k_gemm_wmma<<<dim3(N_NODES / 16, OUT_DIM / 16), 32, 0, stream>>>(out1, W2, h2, C1, OUT_DIM);
    k_attn2<<<g(N_NODES, B), B, 0, stream>>>(h2, as2, ad2, al_s2, al_d2);
    k_fill_u32<<<g(N_NODES, B), B, 0, stream>>>(m2u, FMAP_NEG_INF, N_NODES);
    k_fill_u32<<<g(N_NODES, B), B, 0, stream>>>((unsigned*)Z2, 0u, N_NODES);
    k_fill_u32<<<g((long long)N_NODES * OUT_DIM, B), B, 0, stream>>>((unsigned*)out, 0u, N_NODES * OUT_DIM);
    k_edge_max2<<<g(E_TOT, B), B, 0, stream>>>(ei, al_s2, al_d2, ebuf, m2u);
    k_edge_exp2<<<g(E_TOT, B), B, 0, stream>>>(ei, ebuf, m2u, Z2);
    k_edge_msg2<<<g(E_TOT, B), B, 0, stream>>>(ei, ebuf, Z2, h2, out);
    k_bias2<<<g((long long)N_NODES * OUT_DIM, B), B, 0, stream>>>(out, b2);

    (void)in_sizes; (void)n_in; (void)out_size; (void)ws_size;
}
